// MultiHeadAttention_16071767622397
// MI455X (gfx1250) — matrix operand order repaired and checked by reading, not compiled, not run on hardware
//
#include <hip/hip_runtime.h>
#include <math.h>

// ---------------------------------------------------------------------------
// MHA forward for MI455X (gfx1250, wave32, WMMA, async-to-LDS).
// Pipeline:
//   pack:   x -> f16, W_qkv -> f16 transposed [N][K], W_out -> f16 transposed
//   qkv:    GEMM (f16 WMMA), writes Q/K as [b,h,n,d], V as [b,h,d,n] (pre-T)
//   attn:   flash attention (f16 WMMA, online softmax), K/V staged via
//           global_load_async_to_lds_b128
//   out:    GEMM (f16 WMMA) + bias -> f32
// All GEMM/attention tile staging uses ASYNCcnt-tracked DMA into LDS.
// ---------------------------------------------------------------------------

typedef __attribute__((ext_vector_type(16))) _Float16 v16h;
typedef __attribute__((ext_vector_type(8)))  _Float16 v8h;
typedef __attribute__((ext_vector_type(8)))  float    v8f;
typedef int v4i __attribute__((vector_size(16)));   // matches builtin's V4i

#define NCTX   2048
#define DIM    1024
#define HEADS  16
#define HDIM   64

// workspace layout (in halves)
#define H_XH    0ull                        // x f16        [4096][1024]
#define H_WQKVT (H_XH    + 4194304ull)      // W_qkv^T f16  [3072][1024]
#define H_WOUTT (H_WQKVT + 3145728ull)      // W_out^T f16  [1024][1024]
#define H_Q     (H_WOUTT + 1048576ull)      // Q  [bh][n][d]
#define H_K     (H_Q     + 4194304ull)      // K  [bh][n][d]
#define H_VT    (H_K     + 4194304ull)      // V^T[bh][d][n]
#define H_ATTN  (H_VT    + 4194304ull)      // attn f16 [4096][1024]

// ---------------- async global->LDS copy (16 bytes / lane) -----------------
#if __has_builtin(__builtin_amdgcn_global_load_async_to_lds_b128) && \
    __has_builtin(__builtin_amdgcn_s_wait_asynccnt)
#define USE_ASYNC_LDS 1
typedef __attribute__((address_space(1))) v4i* as1_v4i_ptr;
typedef __attribute__((address_space(3))) v4i* as3_v4i_ptr;
#endif

__device__ inline void async_copy16(const _Float16* g, _Float16* l) {
#ifdef USE_ASYNC_LDS
  __builtin_amdgcn_global_load_async_to_lds_b128(
      (as1_v4i_ptr)g, (as3_v4i_ptr)l, 0, 0);
#else
  *(uint4*)l = *(const uint4*)g;
#endif
}
__device__ inline void async_wait_all() {
#ifdef USE_ASYNC_LDS
  __builtin_amdgcn_s_wait_asynccnt(0);
#endif
}

// ---------------- WMMA fragment loaders (from LDS) -------------------------
// A-frag (16x32 f16): lane L: M=L&15; halves 0..7 = K 8*(L>>4)+j, 8..15 = +16
__device__ inline v16h load_frag_a(const _Float16* p, int ld, int row0, int k0) {
  const int lane = threadIdx.x & 31;
  const int r  = row0 + (lane & 15);
  const int kb = k0 + 8 * (lane >> 4);
  v8h lo = *(const v8h*)(p + r * ld + kb);
  v8h hi = *(const v8h*)(p + r * ld + kb + 16);
  return __builtin_shufflevector(lo, hi, 0,1,2,3,4,5,6,7,8,9,10,11,12,13,14,15);
}
// B-frag (32x16 f16) from N-major LDS: lane L: N=L&15; halves 0..7 = K 8*(L>>4)+j, 8..15 = +16
__device__ inline v16h load_frag_b(const _Float16* p, int ld, int col0, int k0) {
  const int lane = threadIdx.x & 31;
  const int c  = col0 + (lane & 15);
  const int kb = k0 + 8 * (lane >> 4);
  v8h lo = *(const v8h*)(p + c * ld + kb);
  v8h hi = *(const v8h*)(p + c * ld + kb + 16);
  return __builtin_shufflevector(lo, hi, 0,1,2,3,4,5,6,7,8,9,10,11,12,13,14,15);
}

// ---------------------------------------------------------------------------
// Pack kernels
// ---------------------------------------------------------------------------
__global__ __launch_bounds__(256) void cvt_f16(const float* __restrict__ s,
                                               _Float16* __restrict__ d) {
  size_t i = (size_t)blockIdx.x * blockDim.x + threadIdx.x;   // float4 index
  float4 f = ((const float4*)s)[i];
  union { _Float16 h[4]; uint2 u; } p;
  p.h[0] = (_Float16)f.x; p.h[1] = (_Float16)f.y;
  p.h[2] = (_Float16)f.z; p.h[3] = (_Float16)f.w;
  ((uint2*)d)[i] = p.u;
}

// src [K][N] f32 -> dst [N][K] f16 (coalesced read, scattered 2B write)
__global__ __launch_bounds__(256) void cvtT_f16(const float* __restrict__ s,
                                                _Float16* __restrict__ d,
                                                int K, int N) {
  int k = blockIdx.y;
  int n = blockIdx.x * blockDim.x + threadIdx.x;
  if (n < N) d[(size_t)n * K + k] = (_Float16)s[(size_t)k * N + n];
}

// ---------------------------------------------------------------------------
// Kernel 1: qkv = x @ W_qkv (f16 WMMA), scatter to Q/K/V^T sections.
// 256 thr (8 waves), tile 64(M) x 128(N), K-step 32; wave = 32x32 (2x2 WMMA).
// ---------------------------------------------------------------------------
__global__ __launch_bounds__(256) void qkv_gemm(const _Float16* __restrict__ Xh,
                                                const _Float16* __restrict__ Wt,
                                                _Float16* __restrict__ Qw,
                                                _Float16* __restrict__ Kw,
                                                _Float16* __restrict__ Vtw) {
  __shared__ __align__(16) _Float16 As[64 * 40];    // [m][k]
  __shared__ __align__(16) _Float16 Bs[128 * 40];   // [n][k]
  const int tid  = threadIdx.x;
  const int lane = tid & 31;
  const int wave = tid >> 5;
  const int wm = wave >> 2, wn = wave & 3;
  const int rowBase = blockIdx.x * 64;
  const int colBase = blockIdx.y * 128;

  v8f acc[2][2] = {};

  const int ar = tid >> 2, ak = (tid & 3) * 8;      // A: 64 x 32, 8 halves/thr
  const int bn = tid >> 1, bko = (tid & 1) * 16;    // B: 128 x 32, 16 halves/thr

  for (int kk = 0; kk < DIM; kk += 32) {
    async_copy16(Xh + (size_t)(rowBase + ar) * DIM + kk + ak, As + ar * 40 + ak);
    const _Float16* wsrc = Wt + (size_t)(colBase + bn) * DIM + kk + bko;
    async_copy16(wsrc,     Bs + bn * 40 + bko);
    async_copy16(wsrc + 8, Bs + bn * 40 + bko + 8);
    async_wait_all();
    __syncthreads();

    #pragma unroll
    for (int i = 0; i < 2; ++i) {
      v16h a = load_frag_a(As, 40, wm * 32 + i * 16, 0);
      #pragma unroll
      for (int j = 0; j < 2; ++j) {
        v16h b = load_frag_b(Bs, 40, wn * 32 + j * 16, 0);
        acc[i][j] = __builtin_amdgcn_wmma_f32_16x16x32_f16(
            false, a, false, b, (short)0, acc[i][j], false, false);
      }
    }
    __syncthreads();
  }

  const int half = lane >> 4, ln = lane & 15;
  #pragma unroll
  for (int i = 0; i < 2; ++i)
    #pragma unroll
    for (int j = 0; j < 2; ++j)
      #pragma unroll
      for (int r = 0; r < 8; ++r) {
        int m = rowBase + wm * 32 + i * 16 + r + 8 * half;   // b*NCTX + n
        int c = colBase + wn * 32 + j * 16 + ln;             // 0..3071
        int sec = c >> 10;
        int w_  = c & 1023;
        int h = w_ >> 6, d = w_ & 63;
        int b = m >> 11, n = m & 2047;
        size_t bh = (size_t)b * HEADS + h;
        _Float16 hv = (_Float16)acc[i][j][r];
        if (sec == 0)      Qw [bh * NCTX * HDIM + (size_t)n * HDIM + d] = hv;
        else if (sec == 1) Kw [bh * NCTX * HDIM + (size_t)n * HDIM + d] = hv;
        else               Vtw[bh * NCTX * HDIM + (size_t)d * NCTX + n] = hv;
      }
}

// ---------------------------------------------------------------------------
// Kernel 2: flash attention. 128 thr (4 waves), 64-query block per WG.
// Q/K/V^T tiles staged with async-to-LDS (V already transposed in memory).
// ---------------------------------------------------------------------------
__global__ __launch_bounds__(128) void attn_flash(const _Float16* __restrict__ Qw,
                                                  const _Float16* __restrict__ Kw,
                                                  const _Float16* __restrict__ Vtw,
                                                  _Float16* __restrict__ attn) {
  __shared__ __align__(16) _Float16 Qs[64 * 72];       // [q][d]
  __shared__ __align__(16) _Float16 Ks[64 * 72];       // [key][d]
  __shared__ __align__(16) _Float16 Vt[64 * 72];       // [d][key]
  __shared__ __align__(16) _Float16 Ps[4 * 16 * 72];   // per-wave P [16][key]
  const int tid  = threadIdx.x;
  const int lane = tid & 31;
  const int wave = tid >> 5;
  const int qb = blockIdx.x;
  const int bh = blockIdx.y;

  const _Float16* Qg  = Qw  + ((size_t)bh * NCTX + qb * 64) * HDIM;
  const _Float16* Kg  = Kw  + (size_t)bh * NCTX * HDIM;
  const _Float16* VtB = Vtw + (size_t)bh * NCTX * HDIM;   // [d][n]

  const int sr = tid >> 1, so = (tid & 1) * 32;   // staging: row, 32-half chunk
  { // Q block: 4 async b128 per thread
    const _Float16* g = Qg + (size_t)sr * HDIM + so;
    _Float16* l = Qs + sr * 72 + so;
    async_copy16(g,      l);
    async_copy16(g + 8,  l + 8);
    async_copy16(g + 16, l + 16);
    async_copy16(g + 24, l + 24);
  }
  async_wait_all();
  __syncthreads();
  v16h qf[2];
  qf[0] = load_frag_a(Qs, 72, wave * 16, 0);
  qf[1] = load_frag_a(Qs, 72, wave * 16, 32);

  float m_run[8], l_run[8];
  v8f acc_o[4] = {};
  #pragma unroll
  for (int r = 0; r < 8; ++r) { m_run[r] = -INFINITY; l_run[r] = 0.0f; }

  const int half = lane >> 4, ln = lane & 15;

  for (int kb = 0; kb < NCTX / 64; ++kb) {
    __syncthreads();   // previous iteration readers done before overwrite
    { // K block [key][d] and V^T block [d][key]: straight async copies
      const _Float16* gk = Kg + ((size_t)(kb * 64 + sr)) * HDIM + so;
      _Float16* lk = Ks + sr * 72 + so;
      async_copy16(gk,      lk);
      async_copy16(gk + 8,  lk + 8);
      async_copy16(gk + 16, lk + 16);
      async_copy16(gk + 24, lk + 24);
      const _Float16* gv = VtB + (size_t)sr * NCTX + kb * 64 + so;
      _Float16* lv = Vt + sr * 72 + so;
      async_copy16(gv,      lv);
      async_copy16(gv + 8,  lv + 8);
      async_copy16(gv + 16, lv + 16);
      async_copy16(gv + 24, lv + 24);
    }
    async_wait_all();
    __syncthreads();

    // S = Q K^T
    v8f s[4] = {};
    #pragma unroll
    for (int kk = 0; kk < 2; ++kk)
      #pragma unroll
      for (int j = 0; j < 4; ++j) {
        v16h bk = load_frag_b(Ks, 72, j * 16, kk * 32);
        s[j] = __builtin_amdgcn_wmma_f32_16x16x32_f16(
            false, qf[kk], false, bk, (short)0, s[j], false, false);
      }

    // online softmax; row = r + 8*(lane>>4) lives in one 16-lane half
    #pragma unroll
    for (int r = 0; r < 8; ++r) {
      float mx = -INFINITY;
      #pragma unroll
      for (int j = 0; j < 4; ++j) { s[j][r] *= 0.125f; mx = fmaxf(mx, s[j][r]); }
      mx = fmaxf(mx, __shfl_xor(mx, 1, 32));
      mx = fmaxf(mx, __shfl_xor(mx, 2, 32));
      mx = fmaxf(mx, __shfl_xor(mx, 4, 32));
      mx = fmaxf(mx, __shfl_xor(mx, 8, 32));
      float mnew = fmaxf(m_run[r], mx);
      float corr = __expf(m_run[r] - mnew);
      m_run[r] = mnew;
      l_run[r] *= corr;
      #pragma unroll
      for (int j = 0; j < 4; ++j) acc_o[j][r] *= corr;
      float rs = 0.0f;
      #pragma unroll
      for (int j = 0; j < 4; ++j) {
        float p = __expf(s[j][r] - mnew);
        rs += p;
        Ps[(wave * 16 + r + 8 * half) * 72 + j * 16 + ln] = (_Float16)p;
      }
      rs += __shfl_xor(rs, 1, 32);
      rs += __shfl_xor(rs, 2, 32);
      rs += __shfl_xor(rs, 4, 32);
      rs += __shfl_xor(rs, 8, 32);
      l_run[r] += rs;
    }

    // O += P @ V (per-wave LDS RAW: DS in-order within wave)
    #pragma unroll
    for (int kk = 0; kk < 2; ++kk) {
      v16h ap = load_frag_a(Ps + wave * 16 * 72, 72, 0, kk * 32);
      #pragma unroll
      for (int j = 0; j < 4; ++j) {
        v16h bv = load_frag_b(Vt, 72, j * 16, kk * 32);
        acc_o[j] = __builtin_amdgcn_wmma_f32_16x16x32_f16(
            false, ap, false, bv, (short)0, acc_o[j], false, false);
      }
    }
  }

  const int b = bh >> 4, h = bh & 15;
  #pragma unroll
  for (int j = 0; j < 4; ++j)
    #pragma unroll
    for (int r = 0; r < 8; ++r) {
      int q = qb * 64 + wave * 16 + r + 8 * half;
      int c = h * HDIM + j * 16 + ln;
      attn[((size_t)b * NCTX + q) * DIM + c] = (_Float16)(acc_o[j][r] / l_run[r]);
    }
}

// ---------------------------------------------------------------------------
// Kernel 3: out = attn @ W_out + b_out (f32 out); A and B^T staged async.
// ---------------------------------------------------------------------------
__global__ __launch_bounds__(256) void out_gemm(const _Float16* __restrict__ A,
                                                const _Float16* __restrict__ Wt,
                                                const float* __restrict__ bias,
                                                float* __restrict__ out) {
  __shared__ __align__(16) _Float16 As[64 * 40];
  __shared__ __align__(16) _Float16 Bs[128 * 40];
  const int tid  = threadIdx.x;
  const int lane = tid & 31;
  const int wave = tid >> 5;
  const int wm = wave >> 2, wn = wave & 3;
  const int rowBase = blockIdx.x * 64;
  const int colBase = blockIdx.y * 128;

  v8f acc[2][2] = {};

  const int ar = tid >> 2, ak = (tid & 3) * 8;
  const int bn = tid >> 1, bko = (tid & 1) * 16;

  for (int kk = 0; kk < DIM; kk += 32) {
    async_copy16(A + (size_t)(rowBase + ar) * DIM + kk + ak, As + ar * 40 + ak);
    const _Float16* wsrc = Wt + (size_t)(colBase + bn) * DIM + kk + bko;
    async_copy16(wsrc,     Bs + bn * 40 + bko);
    async_copy16(wsrc + 8, Bs + bn * 40 + bko + 8);
    async_wait_all();
    __syncthreads();

    #pragma unroll
    for (int i = 0; i < 2; ++i) {
      v16h a = load_frag_a(As, 40, wm * 32 + i * 16, 0);
      #pragma unroll
      for (int j = 0; j < 2; ++j) {
        v16h b = load_frag_b(Bs, 40, wn * 32 + j * 16, 0);
        acc[i][j] = __builtin_amdgcn_wmma_f32_16x16x32_f16(
            false, a, false, b, (short)0, acc[i][j], false, false);
      }
    }
    __syncthreads();
  }

  const int half = lane >> 4, ln = lane & 15;
  #pragma unroll
  for (int i = 0; i < 2; ++i)
    #pragma unroll
    for (int j = 0; j < 2; ++j)
      #pragma unroll
      for (int r = 0; r < 8; ++r) {
        int m = rowBase + wm * 32 + i * 16 + r + 8 * half;
        int c = colBase + wn * 32 + j * 16 + ln;
        out[(size_t)m * DIM + c] = acc[i][j][r] + bias[c];
      }
}

// ---------------------------------------------------------------------------
extern "C" void kernel_launch(void* const* d_in, const int* in_sizes, int n_in,
                              void* d_out, int out_size, void* d_ws, size_t ws_size,
                              hipStream_t stream) {
  const float* x    = (const float*)d_in[0];
  const float* Wqkv = (const float*)d_in[1];
  const float* Wout = (const float*)d_in[2];
  const float* bout = (const float*)d_in[3];
  float* out = (float*)d_out;

  _Float16* ws = (_Float16*)d_ws;
  _Float16* xh    = ws + H_XH;
  _Float16* wqkvt = ws + H_WQKVT;
  _Float16* woutt = ws + H_WOUTT;
  _Float16* qw    = ws + H_Q;
  _Float16* kw    = ws + H_K;
  _Float16* vtw   = ws + H_VT;
  _Float16* attnh = ws + H_ATTN;

  // pack: x -> f16; W_qkv, W_out -> f16 transposed [N][K]
  cvt_f16 <<<dim3(4096),        256, 0, stream>>>(x, xh);          // 4096*1024/4
  cvtT_f16<<<dim3(12, DIM),     256, 0, stream>>>(Wqkv, wqkvt, DIM, 3 * DIM);
  cvtT_f16<<<dim3(4,  DIM),     256, 0, stream>>>(Wout, woutt, DIM, DIM);

  qkv_gemm  <<<dim3(64, 24),    256, 0, stream>>>(xh, wqkvt, qw, kw, vtw);
  attn_flash<<<dim3(NCTX / 64, 2 * HEADS), 128, 0, stream>>>(qw, kw, vtw, attnh);
  out_gemm  <<<dim3(64, 8),     256, 0, stream>>>(attnh, woutt, bout, out);

  (void)in_sizes; (void)n_in; (void)out_size; (void)ws_size;
}
